// BitNetFusedObservationBlock_43576738185551
// MI455X (gfx1250) — matrix operand order repaired and checked by reading, not compiled, not run on hardware
//
#include <hip/hip_runtime.h>
#include <math.h>

// ---------------------------------------------------------------------------
// Problem dimensions (fixed by the reference)
// ---------------------------------------------------------------------------
#define BATCH 8192
#define D_IN  2048      // DB+DA+DE+DM+DS = 512+256+256+512+512
#define FDIM  2048
#define F2DIM 4096

typedef unsigned short u16;

// ---------------------------------------------------------------------------
// WMMA vector types (CDNA5 / gfx1250, wave32)
// ---------------------------------------------------------------------------
typedef __attribute__((ext_vector_type(16))) __bf16 v16bf;
typedef __attribute__((ext_vector_type(8)))  float  v8f;

union Frag {
    v16bf v;
    uint4 q[2];
};

// ---------------------------------------------------------------------------
// bf16 helpers (round-to-nearest-even)
// ---------------------------------------------------------------------------
__device__ __forceinline__ u16 f2bf(float x) {
    unsigned u = __builtin_bit_cast(unsigned, x);
    unsigned r = u + 0x7FFFu + ((u >> 16) & 1u);
    return (u16)(r >> 16);
}
__device__ __forceinline__ float bf2f(u16 h) {
    unsigned u = ((unsigned)h) << 16;
    return __builtin_bit_cast(float, u);
}
__device__ __forceinline__ float gelu_erf(float x) {
    return 0.5f * x * (1.0f + erff(x * 0.70710678118654752f));
}

// CDNA5 async global->LDS copy (bypasses VGPRs, tracked by ASYNCcnt).
// IOFFSET applies to both the LDS and the global address (ISA 15.18.3).
__device__ __forceinline__ void async_copy_b128(unsigned lds_off, const void* gaddr) {
    asm volatile("global_load_async_to_lds_b128 %0, %1, off"
                 :: "v"(lds_off), "v"(gaddr) : "memory");
}
__device__ __forceinline__ void async_copy_b128_x4(unsigned lds_off, const void* gaddr) {
    asm volatile("global_load_async_to_lds_b128 %0, %1, off\n\t"
                 "global_load_async_to_lds_b128 %0, %1, off offset:16\n\t"
                 "global_load_async_to_lds_b128 %0, %1, off offset:32\n\t"
                 "global_load_async_to_lds_b128 %0, %1, off offset:48"
                 :: "v"(lds_off), "v"(gaddr) : "memory");
}
// ASYNCcnt completes in order: wait<=5 retires the 5 oldest (previous tile's)
// DMAs while the 5 just-issued ones stay in flight.
__device__ __forceinline__ void wait_async5() {
    asm volatile("s_wait_asynccnt 0x5" ::: "memory");
}
__device__ __forceinline__ void wait_async0() {
    asm volatile("s_wait_asynccnt 0x0" ::: "memory");
}

// ---------------------------------------------------------------------------
// 1) Concatenate the 5 fp32 inputs into combined bf16 [BATCH, D_IN]
// ---------------------------------------------------------------------------
__global__ __launch_bounds__(256) void concat_kernel(
    const float* __restrict__ zb, const float* __restrict__ za,
    const float* __restrict__ ze, const float* __restrict__ zm,
    const float* __restrict__ zs, u16* __restrict__ out)
{
    int idx = blockIdx.x * 256 + threadIdx.x;     // over BATCH*D_IN
    int b = idx / D_IN;
    int c = idx - b * D_IN;
    float v;
    if      (c <  512) v = zb[b * 512 + c];
    else if (c <  768) v = za[b * 256 + (c - 512)];
    else if (c < 1024) v = ze[b * 256 + (c - 768)];
    else if (c < 1536) v = zm[b * 512 + (c - 1024)];
    else               v = zs[b * 512 + (c - 1536)];
    out[idx] = f2bf(v);
}

// ---------------------------------------------------------------------------
// 2) BitNet absmean quantization: scale reduction + ternary quantize -> bf16
// ---------------------------------------------------------------------------
__global__ __launch_bounds__(64) void zero_kernel(float* __restrict__ p) {
    if (threadIdx.x < 8) p[threadIdx.x] = 0.0f;
}

__global__ __launch_bounds__(256) void abs_sum_kernel(
    const float* __restrict__ w, int n, float* __restrict__ acc)
{
    __shared__ float s[256];
    float a = 0.0f;
    for (int i = blockIdx.x * 256 + threadIdx.x; i < n; i += gridDim.x * 256)
        a += fabsf(w[i]);
    s[threadIdx.x] = a;
    __syncthreads();
    for (int st = 128; st > 0; st >>= 1) {
        if (threadIdx.x < st) s[threadIdx.x] += s[threadIdx.x + st];
        __syncthreads();
    }
    if (threadIdx.x == 0) atomicAdd(acc, s[0]);
}

__global__ __launch_bounds__(256) void quantize_kernel(
    const float* __restrict__ w, int n, const float* __restrict__ acc,
    float inv_n, u16* __restrict__ out)
{
    float s   = acc[0] * inv_n;                 // absmean scale
    float inv = 1.0f / (s + 1e-5f);
    int i = blockIdx.x * 256 + threadIdx.x;
    if (i < n) {
        float q = rintf(fminf(fmaxf(w[i] * inv, -1.0f), 1.0f)) * s;
        out[i] = f2bf(q);
    }
}

__global__ __launch_bounds__(256) void convert_kernel(
    const float* __restrict__ w, int n, u16* __restrict__ out)
{
    int i = blockIdx.x * 256 + threadIdx.x;
    if (i < n) out[i] = f2bf(w[i]);
}

// ---------------------------------------------------------------------------
// 3) Tiled bf16 WMMA GEMM:  Y[M,N] = epi( X[M,K] * W[N,K]^T + bias[N] )
//    Block = 256 threads = 8 wave32s (2 waveM x 4 waveN)
//    Block tile: 64(M) x 256(N). Wave tile: 32 x 64 via 2x4 accumulators
//    (register reuse: each A frag feeds 4 WMMAs, each B frag feeds 2).
//    Double-buffered LDS tiles filled by GLOBAL_LOAD_ASYNC_TO_LDS_B128:
//    tile t+1's DMA flies underneath tile t's 8 WMMAs (in-order ASYNCcnt,
//    threshold wait <=5). K stepped by 32 -> v_wmma_f32_16x16x32_bf16.
// ---------------------------------------------------------------------------
#define TILE_K 32
#define BLK_M  64
#define BLK_N  256

template <int EPI /*0=none,1=gelu*/, int OUT_BF16 /*1=bf16,0=f32*/>
__global__ __launch_bounds__(256) void gemm_bf16_wmma(
    const u16* __restrict__ X, const u16* __restrict__ W,
    const float* __restrict__ bias, void* __restrict__ Yv,
    int M, int N, int K)
{
    __shared__ __align__(16) u16 sA[2][BLK_M * TILE_K];   //  2 x  4 KB
    __shared__ __align__(16) u16 sW[2][BLK_N * TILE_K];   //  2 x 16 KB

    const int tid   = threadIdx.x;
    const int lane  = tid & 31;
    const int wave  = tid >> 5;
    const int waveM = wave >> 2;          // 0..1  -> 32 rows each
    const int waveN = wave & 3;           // 0..3  -> 64 cols each

    const int rowBlock = blockIdx.y * BLK_M;
    const int colBlock = blockIdx.x * BLK_N;

    // A-matrix (16x32 bf16) lane mapping: lanes 0-15 hold M=lane, K {0..7,16..23};
    // lanes 16-31 hold M=lane-16, K {8..15,24..31}.
    const int koffA = (lane < 16) ? 0 : 8;
    // B-matrix (32x16 bf16) lane mapping: lane = N column; lanes 0-15 hold K {0..7,16..23},
    // lanes 16-31 hold K {8..15,24..31} (same as A).
    const int koffB = (lane < 16) ? 0 : 8;
    const int nLane = lane & 15;

    // Cooperative staging indices: A tile = 64 rows x 32 k (one b128/thread),
    // W tile = 256 rows x 32 k (one row = 4 b128 per thread).
    const int arow = tid >> 2;            // 0..63
    const int aseg = (tid & 3) * 8;       // 0,8,16,24

    // LDS byte offsets for the async-DMA destinations (low 32 bits of a
    // generic LDS pointer are the LDS byte address).
    const unsigned ldsA0 = (unsigned)(size_t)(&sA[0][arow * TILE_K + aseg]);
    const unsigned ldsA1 = (unsigned)(size_t)(&sA[1][arow * TILE_K + aseg]);
    const unsigned ldsW0 = (unsigned)(size_t)(&sW[0][tid * TILE_K]);
    const unsigned ldsW1 = (unsigned)(size_t)(&sW[1][tid * TILE_K]);

    const u16* xBase = X + (size_t)(rowBlock + arow) * K + aseg;
    const u16* wBase = W + (size_t)(colBlock + tid) * K;

    const v8f vzero = {0.f, 0.f, 0.f, 0.f, 0.f, 0.f, 0.f, 0.f};
    v8f acc[2][4] = {{vzero, vzero, vzero, vzero},
                     {vzero, vzero, vzero, vzero}};

    // One compute step on a staged tile pair (8 WMMAs).
    auto compute_tile = [&](const u16* sAb, const u16* sWb) {
        Frag a[2];
#pragma unroll
        for (int mi = 0; mi < 2; ++mi) {
            const u16* ap = sAb + (waveM * 32 + mi * 16 + nLane) * TILE_K + koffA;
            a[mi].q[0] = ((const uint4*)ap)[0];
            a[mi].q[1] = ((const uint4*)(ap + 16))[0];
        }
#pragma unroll
        for (int j = 0; j < 4; ++j) {
            Frag b;
            const u16* wp = sWb + (waveN * 64 + j * 16 + nLane) * TILE_K + koffB;
            b.q[0] = ((const uint4*)wp)[0];
            b.q[1] = ((const uint4*)(wp + 16))[0];
            acc[0][j] = __builtin_amdgcn_wmma_f32_16x16x32_bf16(
                false, a[0].v, false, b.v, (short)0, acc[0][j], false, false);
            acc[1][j] = __builtin_amdgcn_wmma_f32_16x16x32_bf16(
                false, a[1].v, false, b.v, (short)0, acc[1][j], false, false);
        }
    };

    // Prologue: DMA tile 0 into buffer 0 (5 async ops / thread).
    async_copy_b128(ldsA0, xBase);
    async_copy_b128_x4(ldsW0, wBase);

    // K/TILE_K is even for every GEMM in this model (64 or 128 steps),
    // so run a 2-phase ping-pong loop with static LDS addressing.
    for (int k0 = 0; k0 < K; k0 += 2 * TILE_K) {
        // ---- phase 0: compute buf0 (tile k0), DMA buf1 (tile k0+32) ----
        __syncthreads();                       // readers of buf1 (prev iter) done
        {
            const u16* xs = xBase + (k0 + TILE_K);
            const u16* wsrc = wBase + (k0 + TILE_K);
            async_copy_b128(ldsA1, xs);
            async_copy_b128_x4(ldsW1, wsrc);
            __builtin_prefetch(wsrc + TILE_K, 0, 1);
        }
        wait_async5();                         // tile k0 landed (in-order)
        __syncthreads();                       // ... for all waves
        compute_tile(&sA[0][0], &sW[0][0]);

        // ---- phase 1: compute buf1 (tile k0+32), DMA buf0 (tile k0+64) ----
        __syncthreads();                       // readers of buf0 done
        if (k0 + 2 * TILE_K < K) {
            const u16* xs = xBase + (k0 + 2 * TILE_K);
            const u16* wsrc = wBase + (k0 + 2 * TILE_K);
            async_copy_b128(ldsA0, xs);
            async_copy_b128_x4(ldsW0, wsrc);
            __builtin_prefetch(wsrc + TILE_K, 0, 1);
            wait_async5();                     // tile k0+32 landed
        } else {
            wait_async0();                     // final tile: drain everything
        }
        __syncthreads();
        compute_tile(&sA[1][0], &sW[1][0]);
    }

    // C/D layout: VGPR r holds M = r (lanes 0-15) or r+8 (lanes 16-31), N = lane&15
#pragma unroll
    for (int mi = 0; mi < 2; ++mi) {
        const int rowBase = rowBlock + waveM * 32 + mi * 16 + 8 * (lane >> 4);
#pragma unroll
        for (int j = 0; j < 4; ++j) {
            int col = colBlock + waveN * 64 + j * 16 + nLane;
            float bv = bias[col];
#pragma unroll
            for (int r = 0; r < 8; ++r) {
                float v = acc[mi][j][r] + bv;
                if (EPI == 1) v = gelu_erf(v);
                size_t idx = (size_t)(rowBase + r) * (size_t)N + (size_t)col;
                if (OUT_BF16) ((u16*)Yv)[idx] = f2bf(v);
                else          ((float*)Yv)[idx] = v;
            }
        }
    }
}

// ---------------------------------------------------------------------------
// 4) fused = LN1( sigmoid(1.2*g) * t )   (block per row, F = 2048)
// ---------------------------------------------------------------------------
__global__ __launch_bounds__(256) void fuse_ln1_kernel(
    const u16* __restrict__ g, const u16* __restrict__ t,
    const float* __restrict__ lng, const float* __restrict__ lnb,
    u16* __restrict__ out)
{
    __shared__ float s1[256], s2[256];
    const int row = blockIdx.x;
    const int tid = threadIdx.x;

    float v[8];
    float a = 0.0f, b = 0.0f;
#pragma unroll
    for (int i = 0; i < 8; ++i) {
        int col = i * 256 + tid;
        float gv = bf2f(g[(size_t)row * FDIM + col]);
        float tv = bf2f(t[(size_t)row * FDIM + col]);
        float f  = tv / (1.0f + expf(-1.2f * gv));   // sigmoid(1.2*g)*t
        v[i] = f;
        a += f;
        b += f * f;
    }
    s1[tid] = a; s2[tid] = b;
    __syncthreads();
    for (int st = 128; st > 0; st >>= 1) {
        if (tid < st) { s1[tid] += s1[tid + st]; s2[tid] += s2[tid + st]; }
        __syncthreads();
    }
    float mean = s1[0] * (1.0f / FDIM);
    float var  = s2[0] * (1.0f / FDIM) - mean * mean;
    float inv  = rsqrtf(var + 1e-5f);
#pragma unroll
    for (int i = 0; i < 8; ++i) {
        int col = i * 256 + tid;
        out[(size_t)row * FDIM + col] = f2bf((v[i] - mean) * inv * lng[col] + lnb[col]);
    }
}

// ---------------------------------------------------------------------------
// 5) out = LN2( fused + mlp )  -> fp32 output
// ---------------------------------------------------------------------------
__global__ __launch_bounds__(256) void final_ln2_kernel(
    const u16* __restrict__ fused, const float* __restrict__ mlp,
    const float* __restrict__ lng, const float* __restrict__ lnb,
    float* __restrict__ out)
{
    __shared__ float s1[256], s2[256];
    const int row = blockIdx.x;
    const int tid = threadIdx.x;

    float v[8];
    float a = 0.0f, b = 0.0f;
#pragma unroll
    for (int i = 0; i < 8; ++i) {
        int col = i * 256 + tid;
        float f = bf2f(fused[(size_t)row * FDIM + col]) + mlp[(size_t)row * FDIM + col];
        v[i] = f;
        a += f;
        b += f * f;
    }
    s1[tid] = a; s2[tid] = b;
    __syncthreads();
    for (int st = 128; st > 0; st >>= 1) {
        if (tid < st) { s1[tid] += s1[tid + st]; s2[tid] += s2[tid + st]; }
        __syncthreads();
    }
    float mean = s1[0] * (1.0f / FDIM);
    float var  = s2[0] * (1.0f / FDIM) - mean * mean;
    float inv  = rsqrtf(var + 1e-5f);
#pragma unroll
    for (int i = 0; i < 8; ++i) {
        int col = i * 256 + tid;
        out[(size_t)row * FDIM + col] = (v[i] - mean) * inv * lng[col] + lnb[col];
    }
}

// ---------------------------------------------------------------------------
// Host orchestration
// ---------------------------------------------------------------------------
extern "C" void kernel_launch(void* const* d_in, const int* in_sizes, int n_in,
                              void* d_out, int out_size, void* d_ws, size_t ws_size,
                              hipStream_t stream)
{
    const float* z_byte = (const float*)d_in[0];
    const float* z_addr = (const float*)d_in[1];
    const float* z_evt  = (const float*)d_in[2];
    const float* z_map  = (const float*)d_in[3];
    const float* z_sum  = (const float*)d_in[4];
    const float* gate_w1 = (const float*)d_in[5];
    const float* gate_b1 = (const float*)d_in[6];
    const float* gate_w2 = (const float*)d_in[7];
    const float* gate_b2 = (const float*)d_in[8];
    const float* tr_w1   = (const float*)d_in[9];
    const float* tr_b1   = (const float*)d_in[10];
    const float* tr_w2   = (const float*)d_in[11];
    const float* tr_b2   = (const float*)d_in[12];
    const float* mlp_w1  = (const float*)d_in[13];
    const float* mlp_b1  = (const float*)d_in[14];
    const float* mlp_w2  = (const float*)d_in[15];
    const float* mlp_b2  = (const float*)d_in[16];
    const float* ln1_g   = (const float*)d_in[17];
    const float* ln1_b   = (const float*)d_in[18];
    const float* ln2_g   = (const float*)d_in[19];
    const float* ln2_b   = (const float*)d_in[20];

    // Workspace layout (lifetime-aliased), MB offsets:
    //   [0,32)   combined bf16        (steps 1-5)   } overlapped later by
    //   [32,64)  A0 hidden bf16       (steps 3-5)   }   A4 mlp-hidden [0,64)
    //   [64,80)  Q  quantized weights bf16 (transient, every stage)
    //   [80,112) A1 g bf16            (4-6)         } overlapped later by
    //   [112,144)A2 t bf16            (5-6)         }   A5 mlp-out f32 [80,144)
    //   [144,176)A3 fused bf16        (6-9)
    //   [176,+)  scale accumulators
    char* ws = (char*)d_ws;
    const size_t MB = 1ull << 20;
    u16*   C  = (u16*)(ws + 0 * MB);
    u16*   A0 = (u16*)(ws + 32 * MB);
    u16*   A4 = (u16*)(ws + 0 * MB);      // aliases C+A0 (dead by MLP stage)
    u16*   Q  = (u16*)(ws + 64 * MB);
    u16*   A1 = (u16*)(ws + 80 * MB);
    u16*   A2 = (u16*)(ws + 112 * MB);
    float* A5 = (float*)(ws + 80 * MB);   // aliases A1+A2 (dead after fuse)
    u16*   A3 = (u16*)(ws + 144 * MB);
    float* SC = (float*)(ws + 176 * MB);

    const int nW1 = FDIM * D_IN;          // 2048*2048
    const int nW2 = FDIM * FDIM;          // 2048*2048
    const int nM1 = F2DIM * FDIM;         // 4096*2048
    const int nM2 = FDIM * F2DIM;         // 2048*4096

    dim3 blk(256);
    dim3 gridF (FDIM  / BLK_N, BATCH / BLK_M);   // (8, 128)
    dim3 gridF2(F2DIM / BLK_N, BATCH / BLK_M);   // (16, 128)

    // 1) combined = concat(inputs) -> bf16
    concat_kernel<<<(BATCH * D_IN) / 256, blk, 0, stream>>>(
        z_byte, z_addr, z_evt, z_map, z_sum, C);
    zero_kernel<<<1, 64, 0, stream>>>(SC);

    // 2) gate path
    abs_sum_kernel<<<1024, blk, 0, stream>>>(gate_w1, nW1, SC + 0);
    quantize_kernel<<<nW1 / 256, blk, 0, stream>>>(gate_w1, nW1, SC + 0, 1.0f / nW1, Q);
    gemm_bf16_wmma<1, 1><<<gridF, blk, 0, stream>>>(C, Q, gate_b1, A0, BATCH, FDIM, D_IN);

    abs_sum_kernel<<<1024, blk, 0, stream>>>(gate_w2, nW2, SC + 1);
    quantize_kernel<<<nW2 / 256, blk, 0, stream>>>(gate_w2, nW2, SC + 1, 1.0f / nW2, Q);
    gemm_bf16_wmma<0, 1><<<gridF, blk, 0, stream>>>(A0, Q, gate_b2, A1, BATCH, FDIM, FDIM);

    // 3) transform path (reuses A0)
    abs_sum_kernel<<<1024, blk, 0, stream>>>(tr_w1, nW1, SC + 2);
    quantize_kernel<<<nW1 / 256, blk, 0, stream>>>(tr_w1, nW1, SC + 2, 1.0f / nW1, Q);
    gemm_bf16_wmma<1, 1><<<gridF, blk, 0, stream>>>(C, Q, tr_b1, A0, BATCH, FDIM, D_IN);

    abs_sum_kernel<<<1024, blk, 0, stream>>>(tr_w2, nW2, SC + 3);
    quantize_kernel<<<nW2 / 256, blk, 0, stream>>>(tr_w2, nW2, SC + 3, 1.0f / nW2, Q);
    gemm_bf16_wmma<0, 1><<<gridF, blk, 0, stream>>>(A0, Q, tr_b2, A2, BATCH, FDIM, FDIM);

    // 4) fused = LN1(sigmoid(1.2 g) * t)
    fuse_ln1_kernel<<<BATCH, blk, 0, stream>>>(A1, A2, ln1_g, ln1_b, A3);

    // 5) MLP (plain linear; weights just converted to bf16)
    convert_kernel<<<nM1 / 256, blk, 0, stream>>>(mlp_w1, nM1, Q);
    gemm_bf16_wmma<1, 1><<<gridF2, blk, 0, stream>>>(A3, Q, mlp_b1, A4, BATCH, F2DIM, FDIM);

    convert_kernel<<<nM2 / 256, blk, 0, stream>>>(mlp_w2, nM2, Q);
    gemm_bf16_wmma<0, 0><<<gridF, blk, 0, stream>>>(A4, Q, mlp_b2, A5, BATCH, FDIM, F2DIM);

    // 6) out = LN2(fused + mlp)
    final_ln2_kernel<<<BATCH, blk, 0, stream>>>(A3, A5, ln2_g, ln2_b, (float*)d_out);
}
